// HyperbolicGraphConvolution_8770323218936
// MI455X (gfx1250) — matrix operand order repaired and checked by reading, not compiled, not run on hardware
//
#include <hip/hip_runtime.h>
#include <hip/hip_bf16.h>
#include <math.h>

// ---------------------------------------------------------------------------
// Problem constants (match reference)
// ---------------------------------------------------------------------------
#define NN_NODES 50000
#define NE_EDGES 600000
#define NG_GRAPH 512
#define NF 128
#define ND 128
#define NED 384            // D * L
#define SEG ((size_t)NG_GRAPH * NED)      // 196608
#define NSEG ((size_t)NN_NODES * NED)     // 19200000

typedef __attribute__((ext_vector_type(16))) __bf16 v16bf;
typedef __attribute__((ext_vector_type(8)))  __bf16 v8bf;
typedef __attribute__((ext_vector_type(8)))  float  v8f;

// ---------------------------------------------------------------------------
// Utility kernels
// ---------------------------------------------------------------------------
__global__ void fill0_kernel(float* __restrict__ p, long n) {
    long i = (long)blockIdx.x * blockDim.x + threadIdx.x;
    if (i < n) p[i] = 0.0f;
}

__global__ void deg_kernel(const int* __restrict__ dst, float* __restrict__ deg, int E) {
    int e = blockIdx.x * blockDim.x + threadIdx.x;
    if (e < E) atomicAdd(&deg[dst[e]], 1.0f);
}

__global__ void dinv_kernel(const float* __restrict__ deg, float* __restrict__ dinv, int n) {
    int i = blockIdx.x * blockDim.x + threadIdx.x;
    if (i < n) dinv[i] = rsqrtf(deg[i] + 1.0f);
}

// logmap0 on rows of 128 floats, one wave32 per row
__global__ void logmap_kernel(const float* __restrict__ x, float* __restrict__ xt, int rows) {
    long gid = (long)blockIdx.x * blockDim.x + threadIdx.x;
    int row = (int)(gid >> 5);
    int lane = (int)(gid & 31);
    if (row >= rows) return;
    const float* p = x + (size_t)row * NF;
    float v[4], ss = 0.0f;
#pragma unroll
    for (int j = 0; j < 4; ++j) { v[j] = p[lane + 32 * j]; ss += v[j] * v[j]; }
#pragma unroll
    for (int m = 16; m >= 1; m >>= 1) ss += __shfl_xor(ss, m, 32);
    float n = fmaxf(sqrtf(ss), 1e-15f);
    float z = fminf(n, 1.0f - 1e-15f);
    float f = atanhf(z) / n;
    float* q = xt + (size_t)row * NF;
#pragma unroll
    for (int j = 0; j < 4; ++j) q[lane + 32 * j] = v[j] * f;
}

// expmap0 + PoincareBall proj, in place, rows of 384 floats, one wave per row
__global__ void expmap_proj_kernel(float* __restrict__ y, int rows) {
    long gid = (long)blockIdx.x * blockDim.x + threadIdx.x;
    int row = (int)(gid >> 5);
    int lane = (int)(gid & 31);
    if (row >= rows) return;
    float* p = y + (size_t)row * NED;
    float v[12], ss = 0.0f;
#pragma unroll
    for (int j = 0; j < 12; ++j) { v[j] = p[lane + 32 * j]; ss += v[j] * v[j]; }
#pragma unroll
    for (int m = 16; m >= 1; m >>= 1) ss += __shfl_xor(ss, m, 32);
    float n = fmaxf(sqrtf(ss), 1e-15f);
    float t = tanhf(n);
    float f = t / n;                      // expmap0 scale; resulting norm == t
    const float maxn = 1.0f - 4e-3f;
    float tn = fmaxf(t, 1e-15f);
    if (tn > maxn) f *= maxn / tn;        // proj
#pragma unroll
    for (int j = 0; j < 12; ++j) p[lane + 32 * j] = v[j] * f;
}

// agg = h * dinv^2 + b  (self-loop term + bias), over N x 128
__global__ void agg_init_kernel(const float* __restrict__ h, const float* __restrict__ b,
                                const float* __restrict__ dinv, float* __restrict__ agg, int n) {
    long idx = (long)blockIdx.x * blockDim.x + threadIdx.x;
    if (idx >= (long)n * ND) return;
    int i = (int)(idx >> 7);
    int d = (int)(idx & 127);
    float di = dinv[i];
    agg[idx] = h[idx] * di * di + b[d];
}

// one wave per edge; each lane float4-gathers h[src] and atomically adds into agg[dst]
__global__ void edge_scatter_kernel(const int* __restrict__ src, const int* __restrict__ dst,
                                    const float* __restrict__ dinv, const float* __restrict__ h,
                                    float* __restrict__ agg, int E) {
    long tid = (long)blockIdx.x * blockDim.x + threadIdx.x;
    if (tid >= (long)E * 32) return;
    int e = (int)(tid >> 5);
    int q = (int)((tid & 31) << 2);
    int s = src[e], d = dst[e];
    float coef = dinv[s] * dinv[d];
    const float4 hv = *(const float4*)(h + (size_t)s * ND + q);
    float* ap = agg + (size_t)d * ND + q;
    atomicAdd(ap + 0, hv.x * coef);
    atomicAdd(ap + 1, hv.y * coef);
    atomicAdd(ap + 2, hv.z * coef);
    atomicAdd(ap + 3, hv.w * coef);
}

// relu(agg) -> ncat[:, loff:loff+128], and pooled atomicAdd into gpool[batch, poff:poff+128]
__global__ void relu_pool_kernel(const float* __restrict__ agg, float* __restrict__ ncat,
                                 float* __restrict__ gpool, const int* __restrict__ batch,
                                 int n, int loff, int poff) {
    long idx = (long)blockIdx.x * blockDim.x + threadIdx.x;
    if (idx >= (long)n * ND) return;
    int i = (int)(idx >> 7);
    int d = (int)(idx & 127);
    float v = agg[idx];
    v = v > 0.0f ? v : 0.0f;
    ncat[(size_t)i * NED + loff + d] = v;
    atomicAdd(&gpool[(size_t)batch[i] * (2 * NED) + poff + d], v);
}

// ---------------------------------------------------------------------------
// WMMA GEMM helpers
// ---------------------------------------------------------------------------
__device__ __forceinline__ unsigned int pack2_bf16(float x, float y) {
    __bf16 a = (__bf16)x, b = (__bf16)y;
    unsigned short ua = __builtin_bit_cast(unsigned short, a);
    unsigned short ub = __builtin_bit_cast(unsigned short, b);
    return (unsigned int)ua | ((unsigned int)ub << 16);
}

// build a v16bf fragment from two 16-byte-aligned 8-element LDS chunks
__device__ __forceinline__ v16bf load_frag2(const __bf16* p0, const __bf16* p1) {
    v8bf lo = *(const v8bf*)p0;
    v8bf hi = *(const v8bf*)p1;
    v16bf r;
#pragma unroll
    for (int j = 0; j < 8; ++j) { r[j] = lo[j]; r[8 + j] = hi[j]; }
    return r;
}

// ---------------------------------------------------------------------------
// WMMA GEMM: C[M,Nout] = A[M,K] (lda) * B[K,Nout] (ldb), fp32 in/out,
// bf16 A/B fragments, fp32 accumulate.
// Block tile 128x64, 8 waves arranged 4(M) x 2(N); each wave owns a 32x32
// output tile = 2 A-frags x 2 B-frags -> 4 v_wmma per K-step from
// 8 ds_load_b128 (2x the matrix work per LDS byte vs a 16x32 strip).
// B is staged TRANSPOSED in LDS so every fragment is 2x ds_load_b128.
// K must be a multiple of 32; Nout a multiple of 64; M row-clamped.
// ---------------------------------------------------------------------------
__global__ __launch_bounds__(256) void gemm_bf16_wmma_kernel(
    const float* __restrict__ A, int lda,
    const float* __restrict__ B, int ldb,
    float* __restrict__ C, int ldc,
    int M, int K, int relu)
{
    __shared__ __bf16 As[128][40];  // [m][k], 80-byte row stride (16B-aligned frags)
    __shared__ __bf16 Bt[64][40];   // [n][k] transposed, same stride

    const int tid  = threadIdx.x;
    const int lane = tid & 31;
    const int wave = tid >> 5;
    const int mw   = wave & 3;       // M 32-row subtile 0..3
    const int nh   = wave >> 2;      // N half           0..1
    const int blockM = blockIdx.x * 128;
    const int blockN = blockIdx.y * 64;

    const int lrow  = lane & 15;     // m (A) / n (B,C) index within 16
    const int lhalf = lane >> 4;     // lane-half selector

    v8f acc00 = {}, acc01 = {}, acc10 = {}, acc11 = {};

    for (int k0 = 0; k0 < K; k0 += 32) {
        __syncthreads();
        // stage A 128x32 tile (row-clamped): float2 loads, packed bf16x2 LDS stores
#pragma unroll
        for (int i = 0; i < 8; ++i) {
            int e = tid + i * 256;           // pair index, 2048 pairs
            int r = e >> 4;                  // 16 pairs per 32-wide row
            int c = (e & 15) << 1;
            int gr = blockM + r;
            gr = gr < M ? gr : (M - 1);
            const float2 av = *(const float2*)(A + (size_t)gr * lda + k0 + c);
            *(unsigned int*)&As[r][c] = pack2_bf16(av.x, av.y);
        }
        // stage B 32x64 tile transposed: coalesced float2 along n, scalar LDS stores
#pragma unroll
        for (int i = 0; i < 4; ++i) {
            int e = tid + i * 256;           // pair index, 1024 pairs
            int r = e >> 5;                  // k row, 32 pairs per 64-wide row
            int c = (e & 31) << 1;           // n
            const float2 bv = *(const float2*)(B + (size_t)(k0 + r) * ldb + blockN + c);
            Bt[c][r]     = (__bf16)bv.x;
            Bt[c + 1][r] = (__bf16)bv.y;
        }
        __syncthreads();

        // A fragments (16x32 each): lane holds K = {8h..8h+7, 16+8h..+7} of row m
        const __bf16* a0p = &As[mw * 32 + lrow][8 * lhalf];
        const __bf16* a1p = &As[mw * 32 + 16 + lrow][8 * lhalf];
        v16bf a0 = load_frag2(a0p, a0p + 16);
        v16bf a1 = load_frag2(a1p, a1p + 16);

        // B fragments (32x16 each): lane holds K = {8h..8h+7, 16+8h..+7} of column n
        const __bf16* b0p = &Bt[nh * 32 + lrow][8 * lhalf];
        const __bf16* b1p = &Bt[nh * 32 + 16 + lrow][8 * lhalf];
        v16bf b0 = load_frag2(b0p, b0p + 16);
        v16bf b1 = load_frag2(b1p, b1p + 16);

        acc00 = __builtin_amdgcn_wmma_f32_16x16x32_bf16(false, a0, false, b0,
                                                        (short)0, acc00, false, false);
        acc01 = __builtin_amdgcn_wmma_f32_16x16x32_bf16(false, a0, false, b1,
                                                        (short)0, acc01, false, false);
        acc10 = __builtin_amdgcn_wmma_f32_16x16x32_bf16(false, a1, false, b0,
                                                        (short)0, acc10, false, false);
        acc11 = __builtin_amdgcn_wmma_f32_16x16x32_bf16(false, a1, false, b1,
                                                        (short)0, acc11, false, false);
    }

    // epilogue: C/D layout -> row = 8*lhalf + r, col = lrow within each 16x16 tile
    const int rowA = blockM + mw * 32 + 8 * lhalf;       // acc0x rows
    const int rowB = rowA + 16;                          // acc1x rows
    float* cpA = C + (size_t)rowA * ldc + blockN + nh * 32 + lrow;
    float* cpB = C + (size_t)rowB * ldc + blockN + nh * 32 + lrow;
#pragma unroll
    for (int r = 0; r < 8; ++r) {
        if (rowA + r < M) {
            float v0 = acc00[r], v1 = acc01[r];
            if (relu) { v0 = fmaxf(v0, 0.0f); v1 = fmaxf(v1, 0.0f); }
            cpA[(size_t)r * ldc]      = v0;
            cpA[(size_t)r * ldc + 16] = v1;
        }
        if (rowB + r < M) {
            float v0 = acc10[r], v1 = acc11[r];
            if (relu) { v0 = fmaxf(v0, 0.0f); v1 = fmaxf(v1, 0.0f); }
            cpB[(size_t)r * ldc]      = v0;
            cpB[(size_t)r * ldc + 16] = v1;
        }
    }
}

// ---------------------------------------------------------------------------
// Host-side helpers
// ---------------------------------------------------------------------------
static inline int nblk(long n, int t) { return (int)((n + t - 1) / t); }

static void gemm(const float* A, int lda, const float* B, int ldb,
                 float* C, int ldc, int M, int K, int Nout, bool relu,
                 hipStream_t s) {
    dim3 grid((M + 127) / 128, Nout / 64);
    gemm_bf16_wmma_kernel<<<grid, 256, 0, s>>>(A, lda, B, ldb, C, ldc, M, K, relu ? 1 : 0);
}

extern "C" void kernel_launch(void* const* d_in, const int* in_sizes, int n_in,
                              void* d_out, int out_size, void* d_ws, size_t ws_size,
                              hipStream_t stream) {
    const float* x    = (const float*)d_in[0];
    const float* x_s  = (const float*)d_in[1];
    const int*   src  = (const int*)d_in[2];
    const int*   dst  = (const int*)d_in[3];
    const int*   batch= (const int*)d_in[4];
    const float* Wf[3]= {(const float*)d_in[5], (const float*)d_in[6], (const float*)d_in[7]};
    const float* bf_[3]={(const float*)d_in[8], (const float*)d_in[9], (const float*)d_in[10]};
    const float* Ws[3]= {(const float*)d_in[11],(const float*)d_in[12],(const float*)d_in[13]};
    const float* bs_[3]={(const float*)d_in[14],(const float*)d_in[15],(const float*)d_in[16]};
    const float* P1 = (const float*)d_in[17];
    const float* P2 = (const float*)d_in[18];
    const float* G1 = (const float*)d_in[19];
    const float* G2 = (const float*)d_in[20];
    float* out = (float*)d_out;

    // workspace layout (floats)
    float* ws    = (float*)d_ws;
    float* xt    = ws;                                   // N*128
    float* hbuf  = xt   + (size_t)NN_NODES * ND;         // N*128
    float* agg   = hbuf + (size_t)NN_NODES * ND;         // N*128
    float* deg   = agg  + (size_t)NN_NODES * ND;         // N
    float* dinv  = deg  + NN_NODES;                      // N
    float* ncat  = dinv + NN_NODES;                      // N*384
    float* t1    = ncat + (size_t)NN_NODES * NED;        // N*384
    float* gpool = t1   + (size_t)NN_NODES * NED;        // G*768
    float* gtmp  = gpool+ (size_t)NG_GRAPH * 2 * NED;    // G*384

    // shared degree normalization + zero the pooled buffer
    fill0_kernel<<<nblk(NN_NODES, 256), 256, 0, stream>>>(deg, NN_NODES);
    fill0_kernel<<<nblk((long)NG_GRAPH * 2 * NED, 256), 256, 0, stream>>>(gpool, (long)NG_GRAPH * 2 * NED);
    deg_kernel<<<nblk(NE_EDGES, 256), 256, 0, stream>>>(dst, deg, NE_EDGES);
    dinv_kernel<<<nblk(NN_NODES, 256), 256, 0, stream>>>(deg, dinv, NN_NODES);

    for (int br = 0; br < 2; ++br) {
        const float* xin  = br ? x_s : x;
        const float* const* W = br ? Ws  : Wf;
        const float* const* b = br ? bs_ : bf_;
        float* outN = out + (br ? (3 * SEG + NSEG) : (2 * SEG));
        int goff = br * NED;

        logmap_kernel<<<nblk((long)NN_NODES * 32, 256), 256, 0, stream>>>(xin, xt, NN_NODES);

        for (int l = 0; l < 3; ++l) {
            const float* Ain = (l == 0) ? xt : (ncat + (l - 1) * ND);
            int lda = (l == 0) ? NF : NED;
            gemm(Ain, lda, W[l], ND, hbuf, ND, NN_NODES, ND, ND, false, stream);
            agg_init_kernel<<<nblk((long)NN_NODES * ND, 256), 256, 0, stream>>>(hbuf, b[l], dinv, agg, NN_NODES);
            edge_scatter_kernel<<<nblk((long)NE_EDGES * 32, 256), 256, 0, stream>>>(src, dst, dinv, hbuf, agg, NE_EDGES);
            relu_pool_kernel<<<nblk((long)NN_NODES * ND, 256), 256, 0, stream>>>(agg, ncat, gpool, batch,
                                                                                 NN_NODES, l * ND, goff + l * ND);
        }

        // node-level projection head: relu(ncat @ P1) @ P2  -> out segment, then expmap∘proj
        gemm(ncat, NED, P1, NED, t1, NED, NN_NODES, NED, NED, true, stream);
        gemm(t1, NED, P2, NED, outN, NED, NN_NODES, NED, NED, false, stream);
        expmap_proj_kernel<<<nblk((long)NN_NODES * 32, 256), 256, 0, stream>>>(outN, NN_NODES);
    }

    // graph-level heads
    // e(g): relu([g_f|g_s] @ G1) @ G2
    gemm(gpool, 2 * NED, G1, NED, gtmp, NED, NG_GRAPH, 2 * NED, NED, true, stream);
    gemm(gtmp, NED, G2, NED, out, NED, NG_GRAPH, NED, NED, false, stream);
    expmap_proj_kernel<<<nblk((long)NG_GRAPH * 32, 256), 256, 0, stream>>>(out, NG_GRAPH);

    // e(mlp(g_f))
    gemm(gpool, 2 * NED, P1, NED, gtmp, NED, NG_GRAPH, NED, NED, true, stream);
    gemm(gtmp, NED, P2, NED, out + SEG, NED, NG_GRAPH, NED, NED, false, stream);
    expmap_proj_kernel<<<nblk((long)NG_GRAPH * 32, 256), 256, 0, stream>>>(out + SEG, NG_GRAPH);

    // e(mlp(g_s))
    gemm(gpool + NED, 2 * NED, P1, NED, gtmp, NED, NG_GRAPH, NED, NED, true, stream);
    gemm(gtmp, NED, P2, NED, out + 2 * SEG + NSEG, NED, NG_GRAPH, NED, NED, false, stream);
    expmap_proj_kernel<<<nblk((long)NG_GRAPH * 32, 256), 256, 0, stream>>>(out + 2 * SEG + NSEG, NG_GRAPH);
}
